// NetVLAD_86139864089396
// MI455X (gfx1250) — matrix operand order repaired and checked by reading, not compiled, not run on hardware
//
#include <hip/hip_runtime.h>
#include <hip/hip_bf16.h>

typedef __bf16 bf16;
typedef __attribute__((ext_vector_type(16))) __bf16 v16bf;
typedef __attribute__((ext_vector_type(8)))  float  v8f;

#define C_DIM 512
#define P_DIM 1024
#define N_IMG 64
#define K_ALL 66
#define K_PAD 80
#define KC    64
#define EPSF  1e-12f
#define PT    128        // positions per block in GEMM1
#define XS_LD 520        // padded LDS row stride (bf16 elems), 1040B = 65*16B

union Frag32B { v16bf v; uint4 q[2]; };

// A fragment: 16x32 (MxK) bf16. Row-major source, row = lane&15.
// lanes 0-15 hold K = {k0..k0+7, k0+16..k0+23}; lanes 16-31 hold {+8, +24}.
__device__ __forceinline__ v16bf load_a_frag(const bf16* __restrict__ base, int ld,
                                             int k0, int lane) {
  const int l = lane & 15, h = lane >> 4;
  const bf16* p = base + l * ld + k0 + h * 8;
  Frag32B f;
  f.q[0] = *(const uint4*)(p);
  f.q[1] = *(const uint4*)(p + 16);
  return f.v;
}
// B fragment: 32x16 (KxN) bf16, stored [N][K] row-major, col = lane&15.
// lanes 0-15 hold K = {k0..k0+7, k0+16..k0+23}; lanes 16-31 hold {+8, +24}.
__device__ __forceinline__ v16bf load_b_frag(const bf16* __restrict__ base, int ld,
                                             int k0, int lane) {
  const int l = lane & 15, h = lane >> 4;
  const bf16* p = base + l * ld + k0 + h * 8;
  Frag32B f;
  f.q[0] = *(const uint4*)(p);
  f.q[1] = *(const uint4*)(p + 16);
  return f.v;
}

// Async global->LDS copy of one 16B chunk (CDNA5, tracked with ASYNCcnt).
// lds_off: byte address within the workgroup LDS allocation (= low 32 bits of
// the generic pointer).
__device__ __forceinline__ void async_copy_b128(unsigned lds_off,
                                                unsigned long long gaddr) {
  asm volatile("global_load_async_to_lds_b128 %0, %1, off"
               :: "v"(lds_off), "v"(gaddr) : "memory");
}
__device__ __forceinline__ void wait_asynccnt0() {
  asm volatile("s_wait_asynccnt 0x0" ::: "memory");
}

// ---------- Kernel 1: per-position L2 normalize of x, emit bf16 xf[c][p] ----
__global__ void k_normalize(const float* __restrict__ x, bf16* __restrict__ xf) {
  const int n = blockIdx.x >> 2;
  const int p = ((blockIdx.x & 3) << 8) + threadIdx.x;
  const float* xp = x + (size_t)n * C_DIM * P_DIM + p;
  float ss = 0.f;
#pragma unroll 4
  for (int c = 0; c < C_DIM; ++c) { float v = xp[c * P_DIM]; ss += v * v; }
  const float inv = 1.f / fmaxf(sqrtf(ss), EPSF);
  bf16* op = xf + (size_t)n * C_DIM * P_DIM + p;
#pragma unroll 4
  for (int c = 0; c < C_DIM; ++c) { op[c * P_DIM] = (bf16)(xp[c * P_DIM] * inv); }
}

// ---------- Kernel 1b: conv_w -> bf16, zero-padded to K_PAD rows -----------
__global__ void k_convert_w(const float* __restrict__ w, bf16* __restrict__ wb) {
  const int k = blockIdx.x;
  for (int c = threadIdx.x; c < C_DIM; c += blockDim.x)
    wb[k * C_DIM + c] = (k < K_ALL) ? (bf16)w[k * C_DIM + c] : (bf16)0.f;
}

// ---------- Kernel 2: logits GEMM (bf16 WMMA) + bias + softmax -> A bf16 ---
// block = (n, p-tile of 128). Each wave owns 16 positions, all 80 k rows.
__global__ void k_logits_softmax(const bf16* __restrict__ xf, const bf16* __restrict__ wb,
                                 const float* __restrict__ bias, bf16* __restrict__ A) {
  extern __shared__ __align__(16) char smem[];
  bf16*  xs = (bf16*)smem;                       // [PT][XS_LD] transposed xf tile
  float* sb = (float*)(smem + (size_t)PT * XS_LD * 2);  // [K_PAD]
  const int n   = blockIdx.x >> 3;
  const int p0  = (blockIdx.x & 7) * PT;
  const int tid = threadIdx.x;
  if (tid < K_PAD) sb[tid] = (tid < K_ALL) ? bias[tid] : 0.f;

  // stage xf[c][p0..p0+127] into LDS as [p][c] (transpose -> ds stores)
  const bf16* xsrc = xf + (size_t)n * C_DIM * P_DIM + p0;
  for (int idx = tid; idx < PT * C_DIM; idx += 256) {
    const int pl = idx & (PT - 1);
    const int c  = idx >> 7;
    xs[pl * XS_LD + c] = xsrc[c * P_DIM + pl];
  }
  __syncthreads();

  const int wave = tid >> 5, lane = tid & 31;
  const int l = lane & 15, h = lane >> 4;
  const bf16* xrow = xs + (wave * 16) * XS_LD;

  v8f acc[5];
#pragma unroll
  for (int m = 0; m < 5; ++m)
#pragma unroll
    for (int i = 0; i < 8; ++i) acc[m][i] = 0.f;

  for (int c0 = 0; c0 < C_DIM; c0 += 32) {
    const v16bf bfrag = load_b_frag(xrow, XS_LD, c0, lane);
    // Preload all 5 A fragments so the loads clause together and the WMMAs
    // drain the load counter incrementally instead of full-wait per op.
    v16bf afr[5];
#pragma unroll
    for (int m = 0; m < 5; ++m)
      afr[m] = load_a_frag(wb + m * 16 * C_DIM, C_DIM, c0, lane);
#pragma unroll
    for (int m = 0; m < 5; ++m)
      acc[m] = __builtin_amdgcn_wmma_f32_16x16x32_bf16(
          false, afr[m], false, bfrag, (short)0, acc[m], false, false);
  }

  // softmax over k (66 valid rows) for position p = p0 + wave*16 + l.
  // D layout: acc[m][r] -> k = m*16 + r + 8*h, p-col = l.
  float mx = -3.4e38f;
#pragma unroll
  for (int m = 0; m < 5; ++m)
#pragma unroll
    for (int r = 0; r < 8; ++r) {
      const int k = m * 16 + r + 8 * h;
      if (k < K_ALL) { const float v = acc[m][r] + sb[k]; acc[m][r] = v; mx = fmaxf(mx, v); }
    }
  mx = fmaxf(mx, __shfl_xor(mx, 16, 32));   // combine the two lane-halves (same p)
  float s = 0.f;
#pragma unroll
  for (int m = 0; m < 5; ++m)
#pragma unroll
    for (int r = 0; r < 8; ++r) {
      const int k = m * 16 + r + 8 * h;
      const float e = (k < K_ALL) ? __expf(acc[m][r] - mx) : 0.f;
      acc[m][r] = e; s += e;
    }
  s += __shfl_xor(s, 16, 32);
  const float inv = 1.f / s;

  // store A (only k < 64 is consumed downstream), layout [n][k][p]
  bf16* aout = A + (size_t)n * KC * P_DIM + (p0 + wave * 16 + l);
#pragma unroll
  for (int m = 0; m < 4; ++m)
#pragma unroll
    for (int r = 0; r < 8; ++r) {
      const int k = m * 16 + r + 8 * h;
      aout[k * P_DIM] = (bf16)(acc[m][r] * inv);
    }
}

// ---------- Kernel 2b: asum[n][k] = sum_p A[n][k][p] ----------------------
__global__ void k_asum(const bf16* __restrict__ A, float* __restrict__ asum) {
  __shared__ float red[256];
  const int n = blockIdx.x >> 6, k = blockIdx.x & 63;
  const bf16* row = A + ((size_t)(n * KC + k)) * P_DIM;
  float s = 0.f;
  for (int p = threadIdx.x; p < P_DIM; p += 256) s += (float)row[p];
  red[threadIdx.x] = s; __syncthreads();
  for (int off = 128; off > 0; off >>= 1) {
    if (threadIdx.x < off) red[threadIdx.x] += red[threadIdx.x + off];
    __syncthreads();
  }
  if (threadIdx.x == 0) asum[n * KC + k] = red[0];
}

// ---------- Kernel 3: agg GEMM (bf16 WMMA) + centroid sub + intra-L2 -------
// block = (n, k-tile of 16). Each wave owns 64 channels (4 N-tiles).
__global__ void k_vlad(const bf16* __restrict__ A, const bf16* __restrict__ xf,
                       const float* __restrict__ asum, const float* __restrict__ cent,
                       float* __restrict__ out) {
  __shared__ __align__(16) bf16 sA[16 * P_DIM];
  __shared__ float s_part[8][16];
  __shared__ float s_inv[16];
  const int n = blockIdx.x >> 2, kt = blockIdx.x & 3;
  const int tid = threadIdx.x;

  // async DMA of the A tile (contiguous 32KB) into LDS, tracked on ASYNCcnt
  const bf16* asrc = A + ((size_t)(n * KC + kt * 16)) * P_DIM;
  {
    const unsigned lds_base = (unsigned)(uintptr_t)(&sA[0]);
    const unsigned long long gbase = (unsigned long long)(uintptr_t)asrc;
#pragma unroll
    for (int i = 0; i < (16 * P_DIM * 2) / (256 * 16); ++i) {   // 8 chunks/thread
      const int chunk = tid + i * 256;
      async_copy_b128(lds_base + chunk * 16, gbase + (unsigned long long)chunk * 16);
    }
  }
  wait_asynccnt0();
  __syncthreads();

  const int wave = tid >> 5, lane = tid & 31;
  const int l = lane & 15, h = lane >> 4;
  const int cbase = wave * 64;
  const bf16* xbase = xf + (size_t)n * C_DIM * P_DIM;

  v8f acc[4];
#pragma unroll
  for (int t = 0; t < 4; ++t)
#pragma unroll
    for (int i = 0; i < 8; ++i) acc[t][i] = 0.f;

  for (int p0 = 0; p0 < P_DIM; p0 += 32) {
    const v16bf af = load_a_frag(sA, P_DIM, p0, lane);
    v16bf bfr[4];
#pragma unroll
    for (int t = 0; t < 4; ++t)
      bfr[t] = load_b_frag(xbase + (cbase + t * 16) * P_DIM, P_DIM, p0, lane);
#pragma unroll
    for (int t = 0; t < 4; ++t)
      acc[t] = __builtin_amdgcn_wmma_f32_16x16x32_bf16(
          false, af, false, bfr[t], (short)0, acc[t], false, false);
  }

  // vlad = agg - asum[k]*centroid[k][c]; accumulate per-row sum of squares
  float part[8];
#pragma unroll
  for (int r = 0; r < 8; ++r) part[r] = 0.f;
  const int k0 = kt * 16;
#pragma unroll
  for (int r = 0; r < 8; ++r) {
    const int k = k0 + r + 8 * h;
    const float av = asum[n * KC + k];
#pragma unroll
    for (int t = 0; t < 4; ++t) {
      const int c = cbase + t * 16 + l;
      const float v = acc[t][r] - av * cent[k * C_DIM + c];
      acc[t][r] = v;
      part[r] += v * v;
    }
  }
  // reduce over the 16 lanes of each half (xor 1,2,4,8 stay within half)
#pragma unroll
  for (int r = 0; r < 8; ++r) {
    part[r] += __shfl_xor(part[r], 1, 32);
    part[r] += __shfl_xor(part[r], 2, 32);
    part[r] += __shfl_xor(part[r], 4, 32);
    part[r] += __shfl_xor(part[r], 8, 32);
  }
  if (l == 0) {           // lane 0 (rows r) and lane 16 (rows r+8) of each wave
#pragma unroll
    for (int r = 0; r < 8; ++r) s_part[wave][r + 8 * h] = part[r];
  }
  __syncthreads();
  if (tid < 16) {
    float t = 0.f;
#pragma unroll
    for (int w = 0; w < 8; ++w) t += s_part[w][tid];
    s_inv[tid] = 1.f / fmaxf(sqrtf(t), EPSF);
  }
  __syncthreads();

  float* orow = out + (size_t)n * (KC * C_DIM);
#pragma unroll
  for (int r = 0; r < 8; ++r) {
    const float iv = s_inv[r + 8 * h];
#pragma unroll
    for (int t = 0; t < 4; ++t)
      orow[(k0 + r + 8 * h) * C_DIM + cbase + t * 16 + l] = acc[t][r] * iv;
  }
}

// ---------- Kernel 4: final per-image L2 over the flattened 64*512 ---------
__global__ void k_final(float* __restrict__ out) {
  __shared__ float red[256];
  __shared__ float s_inv;
  const int n = blockIdx.x;
  float* row = out + (size_t)n * (KC * C_DIM);
  float s = 0.f;
  for (int i = threadIdx.x; i < KC * C_DIM; i += 256) { const float v = row[i]; s += v * v; }
  red[threadIdx.x] = s; __syncthreads();
  for (int off = 128; off > 0; off >>= 1) {
    if (threadIdx.x < off) red[threadIdx.x] += red[threadIdx.x + off];
    __syncthreads();
  }
  if (threadIdx.x == 0) s_inv = 1.f / fmaxf(sqrtf(red[0]), EPSF);
  __syncthreads();
  for (int i = threadIdx.x; i < KC * C_DIM; i += 256) row[i] *= s_inv;
}

extern "C" void kernel_launch(void* const* d_in, const int* in_sizes, int n_in,
                              void* d_out, int out_size, void* d_ws, size_t ws_size,
                              hipStream_t stream) {
  (void)in_sizes; (void)n_in; (void)out_size; (void)ws_size;
  const float* x      = (const float*)d_in[0];
  const float* conv_w = (const float*)d_in[1];
  const float* conv_b = (const float*)d_in[2];
  const float* cent   = (const float*)d_in[3];
  float* out = (float*)d_out;

  // workspace layout (~72.1 MiB)
  char* ws = (char*)d_ws;
  bf16*  xf     = (bf16*)ws;                                            // 64 MiB
  bf16*  wb     = (bf16*)(ws + (size_t)64 * 1024 * 1024);               // 80 KiB (padded slot 128 KiB)
  bf16*  Abuf   = (bf16*)(ws + (size_t)64 * 1024 * 1024 + 131072);      // 8 MiB
  float* asum_d = (float*)(ws + (size_t)64 * 1024 * 1024 + 131072 + (size_t)8 * 1024 * 1024); // 16 KiB

  k_normalize<<<N_IMG * (P_DIM / 256), 256, 0, stream>>>(x, xf);
  k_convert_w<<<K_PAD, 256, 0, stream>>>(conv_w, wb);

  const size_t sh2 = (size_t)PT * XS_LD * 2 + K_PAD * 4;   // ~130.3 KiB dynamic LDS
  k_logits_softmax<<<N_IMG * (P_DIM / PT), 256, sh2, stream>>>(xf, wb, conv_b, Abuf);

  k_asum<<<N_IMG * KC, 256, 0, stream>>>(Abuf, asum_d);
  k_vlad<<<N_IMG * 4, 256, 0, stream>>>(Abuf, xf, asum_d, cent, out);
  k_final<<<N_IMG, 256, 0, stream>>>(out);
}
